// MultiHeadAttention_23003844837788
// MI455X (gfx1250) — matrix operand order repaired and checked by reading, not compiled, not run on hardware
//
#include <hip/hip_runtime.h>

// ---------------------------------------------------------------------------
// CDNA5 (gfx1250, wave32) fused GQA attention block.
// bf16 WMMA (v_wmma_f32_16x16x32_bf16) for all matmuls, f32 accumulation.
// B tiles via TDM (tensor_load_to_lds, 6-arg form) and A tiles via
// global_load_async_to_lds_b128; sync fallbacks otherwise.
// ---------------------------------------------------------------------------

typedef __attribute__((ext_vector_type(16))) __bf16 bf16x16;
typedef __attribute__((ext_vector_type(8)))  __bf16 bf16x8;
typedef __attribute__((ext_vector_type(4)))  __bf16 bf16x4;
typedef __attribute__((ext_vector_type(8)))  float  f32x8;
typedef __attribute__((ext_vector_type(4)))  unsigned int u32x4;
typedef __attribute__((ext_vector_type(8)))  int    i32x8;
typedef __attribute__((ext_vector_type(4)))  int    i32x4;

#if __has_builtin(__builtin_amdgcn_global_load_async_to_lds_b128) && \
    __has_builtin(__builtin_amdgcn_s_wait_asynccnt)
#define HAVE_ASYNC_LDS 1
#else
#define HAVE_ASYNC_LDS 0
#endif

#if __has_builtin(__builtin_amdgcn_tensor_load_to_lds) && \
    __has_builtin(__builtin_amdgcn_s_wait_tensorcnt)
#define HAVE_TDM 1
#else
#define HAVE_TDM 0
#endif

__device__ __forceinline__ f32x8 zero8() {
    f32x8 z = {0.f, 0.f, 0.f, 0.f, 0.f, 0.f, 0.f, 0.f};
    return z;
}

// Splice two 8-element bf16 chunks into one 16-element WMMA operand.
// A-fragment (16-bit A 16x32 layout): low lanes hold K{0..7} then K{16..23},
// high lanes K{8..15} then K{24..31}. B-fragment (32x16): lane%16 = N column,
// K in the same order as A.
__device__ __forceinline__ bf16x16 ld_ab(const __bf16* p0, const __bf16* p1) {
    bf16x8 lo = *(const bf16x8*)p0;
    bf16x8 hi = *(const bf16x8*)p1;
    bf16x16 r;
#pragma unroll
    for (int i = 0; i < 8; ++i) { r[i] = lo[i]; r[i + 8] = hi[i]; }
    return r;
}

__device__ __forceinline__ f32x8 wmma_bf16(bf16x16 a, bf16x16 b, f32x8 c) {
    return __builtin_amdgcn_wmma_f32_16x16x32_bf16(
        false, a, false, b, (short)0, c, false, false);
}

// 16-byte global->LDS copy: async data mover when available, sync otherwise.
__device__ __forceinline__ void copy16_g2l(const __bf16* g, __bf16* l) {
#if HAVE_ASYNC_LDS
    __builtin_amdgcn_global_load_async_to_lds_b128(
        (__attribute__((address_space(1))) i32x4*)const_cast<__bf16*>(g),
        (__attribute__((address_space(3))) i32x4*)l, 0, 0);
#else
    *(bf16x8*)l = *(const bf16x8*)g;
#endif
}

__device__ __forceinline__ void wait_async_copies() {
#if HAVE_ASYNC_LDS
    __builtin_amdgcn_s_wait_asynccnt(0);
#endif
}

__device__ __forceinline__ unsigned int lds_byte_off(const void* p) {
    return (unsigned int)(size_t)(const __attribute__((address_space(3))) void*)p;
}

#if HAVE_TDM
// TDM descriptor for a 2D bf16 tile load (D# groups per ISA 08_async_tensor
// sections 8.3/8.4): tile_h x tile_w elements, row stride = `stride` elems.
__device__ __forceinline__ void tdm_load_tile_bf16(
    const __bf16* gsrc, unsigned int ldsOff,
    int tileW, int tileH, int stride, int tenW, int tenH) {
    unsigned long long ga = (unsigned long long)(size_t)gsrc;
    u32x4 g0;
    g0[0] = 1u;                                     // count=1, user descriptor
    g0[1] = ldsOff;                                 // lds_addr (bytes)
    g0[2] = (unsigned int)(ga & 0xFFFFFFFFu);       // global_addr[31:0]
    g0[3] = (unsigned int)((ga >> 32) & 0x1FFFFFFu) // global_addr[56:32]
          | (2u << 30);                             // type = 2 ("image")
    i32x8 g1;
    g1[0] = (1 << 16);                              // data_size = 1 (2 bytes)
    g1[1] = (int)(((unsigned)tenW & 0xFFFFu) << 16);          // tensor_dim0 lo16
    g1[2] = (int)((((unsigned)tenW >> 16) & 0xFFFFu)
          | (((unsigned)tenH & 0xFFFFu) << 16));              // dim0 hi | dim1 lo
    g1[3] = (int)((((unsigned)tenH >> 16) & 0xFFFFu)
          | (((unsigned)tileW & 0xFFFFu) << 16));             // dim1 hi | tile_dim0
    g1[4] = (int)((unsigned)tileH & 0xFFFFu);                 // tile_dim1 (tile_dim2=0)
    g1[5] = stride;                                            // tensor_dim0_stride lo32
    g1[6] = 0;
    g1[7] = 0;
    i32x4 z4 = {0, 0, 0, 0};
    i32x8 z8 = {0, 0, 0, 0, 0, 0, 0, 0};
    __builtin_amdgcn_tensor_load_to_lds(g0, g1, z4, z4, z8, 0);
}
#endif

// ---------------------------------------------------------------------------
// Elementwise f32 -> bf16 (x pre-conversion).
// ---------------------------------------------------------------------------
__global__ __launch_bounds__(256)
void convert_f32_bf16(const float* __restrict__ in, __bf16* __restrict__ out,
                      size_t n) {
    size_t i = ((size_t)blockIdx.x * blockDim.x + threadIdx.x) * 8;
    if (i >= n) return;
    float4 a = *(const float4*)(in + i);
    float4 b = *(const float4*)(in + i + 4);
    bf16x8 h;
    h[0] = (__bf16)a.x; h[1] = (__bf16)a.y; h[2] = (__bf16)a.z; h[3] = (__bf16)a.w;
    h[4] = (__bf16)b.x; h[5] = (__bf16)b.y; h[6] = (__bf16)b.z; h[7] = (__bf16)b.w;
    *(bf16x8*)(out + i) = h;
}

// ---------------------------------------------------------------------------
// W [N][K] f32 -> WT [K][N] bf16 (tiled transpose, coalesced both sides).
// ---------------------------------------------------------------------------
__global__ __launch_bounds__(256)
void transpose_w_bf16(const float* __restrict__ W, __bf16* __restrict__ WT,
                      int N, int K) {
    __shared__ __bf16 tile[32][33];
    const int k0 = blockIdx.x * 32;
    const int n0 = blockIdx.y * 32;
    const int tx = threadIdx.x & 31;
    const int ty = threadIdx.x >> 5;          // 0..7
#pragma unroll
    for (int i = 0; i < 4; ++i) {
        int n = ty + i * 8;
        tile[n][tx] = (__bf16)W[(size_t)(n0 + n) * K + k0 + tx];
    }
    __syncthreads();
#pragma unroll
    for (int i = 0; i < 4; ++i) {
        int k = ty + i * 8;
        WT[(size_t)(k0 + k) * N + n0 + tx] = tile[tx][k];
    }
}

// ---------------------------------------------------------------------------
// GEMM: Out[m][n] = sum_k A[m][k] * BT[k][n], BT bf16 pre-transposed.
// A is bf16 (A_BF16=true: async/TDM copies) or f32 (register convert path).
// Tile 128x128x32, 8 waves, each wave a 32x64 C sub-tile (2x4 WMMA tiles).
// LDS double-buffered: tile k+1 copies overlap tile k WMMAs.
// ---------------------------------------------------------------------------
template <bool A_BF16>
__global__ __launch_bounds__(256)
void gemm_core(const void* __restrict__ Aany, const __bf16* __restrict__ BT,
               float* __restrict__ Out, int M, int N, int K) {
    __shared__ alignas(16) __bf16 sA[2][128 * 32];   // [m][k], 2 x 8 KB
    __shared__ alignas(16) __bf16 sB[2][32 * 128];   // [k][n], 2 x 8 KB

    const int tid  = threadIdx.x;
    const int lane = tid & 31;
    const int wave = tid >> 5;
    const int wm   = wave & 3;
    const int wn   = wave >> 2;
    const int n0   = blockIdx.x * 128;
    const int m0   = blockIdx.y * 128;

    const bool laneLow = (lane < 16);
    const int  mrow    = lane & 15;
    const int  c0      = laneLow ? 0 : 8;

    const __bf16* Ab = (const __bf16*)Aany;
    const float*  Af = (const float*)Aany;

    f32x8 acc[2][4];
#pragma unroll
    for (int mt = 0; mt < 2; ++mt)
#pragma unroll
        for (int nt = 0; nt < 4; ++nt) acc[mt][nt] = zero8();

    float4 ra[4];

    // ---- stage issue: start global->LDS movement for one K-slab ----
    auto stageA_issue = [&](int buf, int k0) {
        if (A_BF16) {
#pragma unroll
            for (int i = 0; i < 2; ++i) {
                int c = tid + i * 256;               // 512 x 16B chunks
                int row = c >> 2, off = (c & 3) * 8;
                copy16_g2l(Ab + (size_t)(m0 + row) * K + k0 + off,
                           &sA[buf][row * 32 + off]);
            }
        } else {
#pragma unroll
            for (int i = 0; i < 4; ++i) {
                int f = tid + i * 256;               // 1024 float4 slots
                int row = f >> 3, kc4 = (f & 7) << 2;
                ra[i] = *(const float4*)(Af + (size_t)(m0 + row) * K + k0 + kc4);
            }
            __builtin_prefetch(Af + (size_t)(m0 + (tid >> 1)) * K + k0 + 32, 0, 1);
        }
    };
    auto stageB_issue = [&](int buf, int k0) {
#if HAVE_TDM
        if (wave == 0) {
            tdm_load_tile_bf16(BT + (size_t)k0 * N + n0,
                               lds_byte_off(&sB[buf][0]),
                               128, 32, N, N, K);
        }
#else
#pragma unroll
        for (int i = 0; i < 2; ++i) {
            int c = tid + i * 256;                   // 512 x 16B chunks
            int row = c >> 4, col = (c & 15) * 8;
            copy16_g2l(BT + (size_t)(k0 + row) * N + n0 + col,
                       &sB[buf][row * 128 + col]);
        }
#endif
    };
    // ---- stage commit: f32 path converts registers into LDS ----
    auto stageA_commit = [&](int buf) {
        if (!A_BF16) {
#pragma unroll
            for (int i = 0; i < 4; ++i) {
                int f = tid + i * 256;
                int row = f >> 3, kc4 = (f & 7) << 2;
                bf16x4 h;
                h[0] = (__bf16)ra[i].x; h[1] = (__bf16)ra[i].y;
                h[2] = (__bf16)ra[i].z; h[3] = (__bf16)ra[i].w;
                *(bf16x4*)(&sA[buf][row * 32 + kc4]) = h;
            }
        }
    };
    auto wait_copies = [&]() {
#if HAVE_TDM
        __builtin_amdgcn_s_wait_tensorcnt(0);
#endif
        wait_async_copies();
    };

    // Prologue: fill buffer 0.
    stageA_issue(0, 0);
    stageB_issue(0, 0);
    stageA_commit(0);
    wait_copies();
    __syncthreads();

    const int nsteps = K / 32;
    for (int s = 0; s < nsteps; ++s) {
        const int  cur  = s & 1;
        const bool more = (s + 1 < nsteps);
        if (more) {                      // overlap next-slab movement w/ WMMA
            stageA_issue(cur ^ 1, (s + 1) * 32);
            stageB_issue(cur ^ 1, (s + 1) * 32);
        }

        bf16x16 afrag[2];
#pragma unroll
        for (int mt = 0; mt < 2; ++mt) {
            const __bf16* p = &sA[cur][(wm * 32 + mt * 16 + mrow) * 32];
            afrag[mt] = ld_ab(p + c0, p + c0 + 16);
        }
        bf16x16 bfrag[4];
#pragma unroll
        for (int nt = 0; nt < 4; ++nt) {
            const __bf16* p = &sB[cur][wn * 64 + nt * 16 + mrow];
            bf16x16 bb;
#pragma unroll
            for (int i = 0; i < 8; ++i) {
                bb[i]     = p[(c0 + i) * 128];
                bb[i + 8] = p[(c0 + 16 + i) * 128];
            }
            bfrag[nt] = bb;
        }
#pragma unroll
        for (int mt = 0; mt < 2; ++mt)
#pragma unroll
            for (int nt = 0; nt < 4; ++nt)
                acc[mt][nt] = wmma_bf16(afrag[mt], bfrag[nt], acc[mt][nt]);

        if (more) {
            stageA_commit(cur ^ 1);
            wait_copies();
        }
        __syncthreads();
    }

    // 16x16 f32 C layout: VGPR r, lanes 0-15 -> (M=r,N=lane), 16-31 -> (M=r+8).
#pragma unroll
    for (int mt = 0; mt < 2; ++mt)
#pragma unroll
        for (int nt = 0; nt < 4; ++nt)
#pragma unroll
            for (int r = 0; r < 8; ++r) {
                int m = m0 + wm * 32 + mt * 16 + r + (laneLow ? 0 : 8);
                int n = n0 + wn * 64 + nt * 16 + mrow;
                Out[(size_t)m * N + n] = acc[mt][nt][r];
            }
}

// ---------------------------------------------------------------------------
// Per-head RMSNorm + RoPE, in place on Q [B,S,16,64] and K [B,S,4,64].
// ---------------------------------------------------------------------------
__global__ __launch_bounds__(256)
void rmsnorm_rope(float* __restrict__ Q, float* __restrict__ Kv,
                  const float* __restrict__ qg, const float* __restrict__ kg,
                  int S, int total) {
    int row = blockIdx.x * blockDim.x + threadIdx.x;
    if (row >= total) return;
    int hh = row % 20;
    int bs = row / 20;
    int s  = bs % S;

    float* ptr;
    const float* g;
    if (hh < 16) { ptr = Q  + ((size_t)bs * 16 + hh) * 64;        g = qg; }
    else         { ptr = Kv + ((size_t)bs * 4  + (hh - 16)) * 64; g = kg; }

    float v[64];
    float ss = 0.f;
#pragma unroll
    for (int i = 0; i < 16; ++i) {
        float4 t = *(const float4*)(ptr + 4 * i);
        v[4*i+0] = t.x; v[4*i+1] = t.y; v[4*i+2] = t.z; v[4*i+3] = t.w;
        ss += t.x*t.x + t.y*t.y + t.z*t.z + t.w*t.w;
    }
    float inv = rsqrtf(ss * (1.f / 64.f) + 1.1920928955078125e-07f);
#pragma unroll
    for (int i = 0; i < 32; ++i) {
        float x0 = v[2*i]     * inv * g[2*i];
        float x1 = v[2*i + 1] * inv * g[2*i + 1];
        float th  = __expf(-0.28782314f * (float)i);   // 10000^(-2i/64)
        float ang = (float)s * th;
        float sn, cs;
        __sincosf(ang, &sn, &cs);
        ptr[2*i]     = x0 * cs - x1 * sn;
        ptr[2*i + 1] = x1 * cs + x0 * sn;
    }
}

// ---------------------------------------------------------------------------
// Flash attention (causal, GQA 4:1). 128 threads = 4 waves; each wave owns a
// 16-row Q tile; waves share LDS-staged 32-key K(transposed)/V chunks.
// ---------------------------------------------------------------------------
__global__ __launch_bounds__(128)
void flash_attn(const float* Q, const float* __restrict__ K,
                const float* __restrict__ V, float* O, int S) {
    __shared__ alignas(16) __bf16 sKT[64 * 32];        // [d][kv]
    __shared__ alignas(16) __bf16 sV [32 * 64];        // [kv][d]
    __shared__ alignas(16) __bf16 sP [4][16 * 32];     // per-wave P tile

    const int tid  = threadIdx.x;
    const int lane = tid & 31;
    const int wave = tid >> 5;
    const int h    = blockIdx.y;
    const int b    = blockIdx.z;
    const int hkv  = h >> 2;
    const int q0   = (blockIdx.x * 4 + wave) * 16;

    const bool laneLow = (lane < 16);
    const int  mrow    = lane & 15;
    const int  c0      = laneLow ? 0 : 8;

    bf16x16 qa[2];
    {
        const float* qrow = Q + (((size_t)b * S + q0 + mrow) * 16 + h) * 64;
#pragma unroll
        for (int ks = 0; ks < 2; ++ks) {
            bf16x16 a;
#pragma unroll
            for (int j = 0; j < 8; ++j) {
                a[j]     = (__bf16)(qrow[ks * 32 + c0 + j]      * 0.125f);
                a[j + 8] = (__bf16)(qrow[ks * 32 + c0 + 16 + j] * 0.125f);
            }
            qa[ks] = a;
        }
    }

    float mstate[8], lstate[8];
    f32x8 oacc[4];
#pragma unroll
    for (int r = 0; r < 8; ++r) { mstate[r] = -1e30f; lstate[r] = 0.f; }
#pragma unroll
    for (int dt = 0; dt < 4; ++dt) oacc[dt] = zero8();

    const int myLimit    = q0 + 16;
    const int blockLimit = blockIdx.x * 64 + 64;

    for (int kv0 = 0; kv0 < blockLimit; kv0 += 32) {
        __syncthreads();
#pragma unroll
        for (int i = 0; i < 4; ++i) {
            int f  = tid + i * 128;
            int kv = f >> 4;
            int d4 = (f & 15) << 2;
            size_t base = (((size_t)b * S + kv0 + kv) * 4 + hkv) * 64 + d4;
            float4 kq = *(const float4*)(K + base);
            float4 vq = *(const float4*)(V + base);
            sKT[(d4 + 0) * 32 + kv] = (__bf16)kq.x;
            sKT[(d4 + 1) * 32 + kv] = (__bf16)kq.y;
            sKT[(d4 + 2) * 32 + kv] = (__bf16)kq.z;
            sKT[(d4 + 3) * 32 + kv] = (__bf16)kq.w;
            bf16x4 hv;
            hv[0] = (__bf16)vq.x; hv[1] = (__bf16)vq.y;
            hv[2] = (__bf16)vq.z; hv[3] = (__bf16)vq.w;
            *(bf16x4*)(&sV[kv * 64 + d4]) = hv;
        }
        __syncthreads();

        if (kv0 >= myLimit) continue;   // wave-uniform; barriers stay aligned

        f32x8 sc[2];
#pragma unroll
        for (int nt = 0; nt < 2; ++nt) {
            const __bf16* pb = &sKT[nt * 16 + mrow];
            bf16x16 b0, b1;
#pragma unroll
            for (int i = 0; i < 8; ++i) {
                b0[i]     = pb[(c0 + i) * 32];
                b0[i + 8] = pb[(c0 + 16 + i) * 32];
                b1[i]     = pb[(32 + c0 + i) * 32];
                b1[i + 8] = pb[(48 + c0 + i) * 32];
            }
            f32x8 s = zero8();
            s = wmma_bf16(qa[0], b0, s);
            s = wmma_bf16(qa[1], b1, s);
            sc[nt] = s;
        }

        float rowmax[8];
#pragma unroll
        for (int r = 0; r < 8; ++r) {
            int q  = q0 + r + (laneLow ? 0 : 8);
            float s0 = (kv0 + mrow      <= q) ? sc[0][r] : -1e30f;
            float s1 = (kv0 + 16 + mrow <= q) ? sc[1][r] : -1e30f;
            sc[0][r] = s0; sc[1][r] = s1;
            float mx = fmaxf(s0, s1);
            mx = fmaxf(mx, __shfl_xor(mx, 1, 16));
            mx = fmaxf(mx, __shfl_xor(mx, 2, 16));
            mx = fmaxf(mx, __shfl_xor(mx, 4, 16));
            mx = fmaxf(mx, __shfl_xor(mx, 8, 16));
            rowmax[r] = mx;
        }

#pragma unroll
        for (int r = 0; r < 8; ++r) {
            float newm  = fmaxf(mstate[r], rowmax[r]);
            float alpha = __expf(mstate[r] - newm);
            mstate[r]   = newm;
            float p0 = __expf(sc[0][r] - newm);
            float p1 = __expf(sc[1][r] - newm);
            int m = r + (laneLow ? 0 : 8);
            sP[wave][m * 32 + mrow]      = (__bf16)p0;
            sP[wave][m * 32 + 16 + mrow] = (__bf16)p1;
            float rs = p0 + p1;
            rs += __shfl_xor(rs, 1, 16);
            rs += __shfl_xor(rs, 2, 16);
            rs += __shfl_xor(rs, 4, 16);
            rs += __shfl_xor(rs, 8, 16);
            lstate[r] = lstate[r] * alpha + rs;
#pragma unroll
            for (int dt = 0; dt < 4; ++dt) oacc[dt][r] *= alpha;
        }

        {
            const __bf16* pp = &sP[wave][mrow * 32];
            bf16x16 pA = ld_ab(pp + c0, pp + c0 + 16);
#pragma unroll
            for (int dt = 0; dt < 4; ++dt) {
                const __bf16* pv = &sV[dt * 16 + mrow];
                bf16x16 vb;
#pragma unroll
                for (int i = 0; i < 8; ++i) {
                    vb[i]     = pv[(c0 + i) * 64];
                    vb[i + 8] = pv[(c0 + 16 + i) * 64];
                }
                oacc[dt] = wmma_bf16(pA, vb, oacc[dt]);
            }
        }
    }

#pragma unroll
    for (int r = 0; r < 8; ++r) {
        float inv = 1.f / lstate[r];
        int m = r + (laneLow ? 0 : 8);
        size_t base = (((size_t)b * S + q0 + m) * 16 + h) * 64 + mrow;
#pragma unroll
        for (int dt = 0; dt < 4; ++dt)
            O[base + dt * 16] = oacc[dt][r] * inv;
    }
}

// ---------------------------------------------------------------------------
extern "C" void kernel_launch(void* const* d_in, const int* in_sizes, int n_in,
                              void* d_out, int out_size, void* d_ws, size_t ws_size,
                              hipStream_t stream) {
    (void)in_sizes; (void)n_in; (void)out_size; (void)ws_size;

    const float* x  = (const float*)d_in[0];
    const float* wq = (const float*)d_in[1];
    const float* wk = (const float*)d_in[2];
    const float* wv = (const float*)d_in[3];
    const float* wo = (const float*)d_in[4];
    const float* qg = (const float*)d_in[5];
    const float* kg = (const float*)d_in[6];
    float* out = (float*)d_out;

    const int B = 4, S = 2048, D = 2048;
    const int M = B * S;                                 // 8192

    // Workspace (148 MB): xbf | wqT | wkT | wvT | woT (bf16) | Q | K | V (f32)
    __bf16* xbf = (__bf16*)d_ws;
    __bf16* wqT = xbf + (size_t)M * 2048;
    __bf16* wkT = wqT + (size_t)2048 * 2048;
    __bf16* wvT = wkT + (size_t)2048 * 512;
    __bf16* woT = wvT + (size_t)2048 * 512;
    float*  Qw  = (float*)(woT + (size_t)2048 * 2048);
    float*  Kw  = Qw + (size_t)M * 2048;
    float*  Vw  = Kw + (size_t)M * 512;

    dim3 blk(256);
    size_t nx = (size_t)M * D;
    convert_f32_bf16<<<dim3((unsigned)(nx / 2048)), blk, 0, stream>>>(x, xbf, nx);
    transpose_w_bf16<<<dim3(D / 32, 2048 / 32), blk, 0, stream>>>(wq, wqT, 2048, D);
    transpose_w_bf16<<<dim3(D / 32, 512  / 32), blk, 0, stream>>>(wk, wkT, 512,  D);
    transpose_w_bf16<<<dim3(D / 32, 512  / 32), blk, 0, stream>>>(wv, wvT, 512,  D);
    transpose_w_bf16<<<dim3(D / 32, 2048 / 32), blk, 0, stream>>>(wo, woT, 2048, D);

    gemm_core<true ><<<dim3(2048 / 128, M / 128), blk, 0, stream>>>(xbf, wqT, Qw, M, 2048, D);
    gemm_core<true ><<<dim3(512  / 128, M / 128), blk, 0, stream>>>(xbf, wkT, Kw, M, 512,  D);
    gemm_core<true ><<<dim3(512  / 128, M / 128), blk, 0, stream>>>(xbf, wvT, Vw, M, 512,  D);

    rmsnorm_rope<<<dim3((M * 20) / 256), blk, 0, stream>>>(Qw, Kw, qg, kg, S, M * 20);

    flash_attn<<<dim3(S / 64, 16, B), dim3(128), 0, stream>>>(Qw, Kw, Vw, Qw, S);

    gemm_core<false><<<dim3(2048 / 128, M / 128), blk, 0, stream>>>(Qw, woT, out, M, 2048, D);
}
